// EVL_40810779247531
// MI455X (gfx1250) — matrix operand order repaired and checked by reading, not compiled, not run on hardware
//
#include <hip/hip_runtime.h>
#include <math.h>

typedef _Float16 h16;
typedef __attribute__((ext_vector_type(16))) _Float16 v16h;
typedef __attribute__((ext_vector_type(8)))  _Float16 v8h;
typedef __attribute__((ext_vector_type(8)))  float    v8f;
typedef __attribute__((ext_vector_type(4)))  float    v4f;

#define NBATCH 16
#define TDIM   8
#define LDIM   197
#define QV     196      // valid spatial tokens
#define HEADS  12
#define DDIM   64
#define CDIM   768
#define NT     112      // (N * 7) batch-time rows per half
#define KPAD   224      // padded K for second GEMM (7 * 32)
#define CBLK   64       // columns per block in output kernel
#define WKPAD  232      // padded LDS k-stride (16B-aligned rows)

static __device__ __forceinline__ v16h pack16(v4f a0, v4f a1, v4f a2, v4f a3, float scl){
  v16h r;
#pragma unroll
  for (int i = 0; i < 4; ++i){
    r[i]      = (_Float16)(a0[i] * scl);
    r[4 + i]  = (_Float16)(a1[i] * scl);
    r[8 + i]  = (_Float16)(a2[i] * scl);
    r[12 + i] = (_Float16)(a3[i] * scl);
  }
  return r;
}

// A-fragment (16x32 f16), lane m = lane&15, hi = lane>>4:
// a[i] = row[d0 + hi*16 + i], i = 0..15 (contiguous)
static __device__ __forceinline__ v16h ldA_f32(const float* rowp, int d0, int hi, float scl){
  const float* p = rowp + d0 + hi * 16;
  v4f x0 = *(const v4f*)(p);
  v4f x1 = *(const v4f*)(p + 4);
  v4f y0 = *(const v4f*)(p + 8);
  v4f y1 = *(const v4f*)(p + 12);
  return pack16(x0, x1, y0, y1, scl);
}

// B-fragment (32x16 f16), lane n = lane&15, hi = lane>>4:
// b[i] = row[d0 + hi*16 + i], i = 0..15 (contiguous)
static __device__ __forceinline__ v16h ldB_f32(const float* rowp, int d0, int hi){
  const float* p = rowp + d0 + hi * 16;
  v4f x0 = *(const v4f*)(p);
  v4f x1 = *(const v4f*)(p + 4);
  v4f x2 = *(const v4f*)(p + 8);
  v4f x3 = *(const v4f*)(p + 12);
  return pack16(x0, x1, x2, x3, 1.0f);
}

// ---------------------------------------------------------------------------
// Kernel 1: head-averaged softmax(QK^T/sqrt(d)) probabilities -> f16 workspace
// grid (13 q-tiles, 14 = t'(7) x half(2), 16 = n), 1 wave per block
// ---------------------------------------------------------------------------
__global__ __launch_bounds__(32)
void EVL_attn_probs(const float* __restrict__ q, const float* __restrict__ k,
                    h16* __restrict__ pws){
  const int qt_i = blockIdx.x;          // q tile: 0..12
  const int tph  = blockIdx.y;          // 0..13
  const int n    = blockIdx.z;
  const int tp   = tph % 7;
  const int hf   = tph / 7;
  const int qt   = hf ? tp : tp + 1;    // half0: q from t'+1 ; half1: q from t'
  const int kt   = hf ? tp + 1 : tp;    // half0: k from t'   ; half1: k from t'+1
  const int lane = threadIdx.x;
  const int m15  = lane & 15;
  const int hi   = lane >> 4;

  __shared__ float sS[16][208];
  __shared__ float pacc[16][KPAD];
  for (int i = lane; i < 16 * KPAD; i += 32) (&pacc[0][0])[i] = 0.0f;

  int mrow = qt_i * 16 + m15;
  if (mrow > QV - 1) mrow = QV - 1;     // clamp pad rows (never stored)

  for (int h = 0; h < HEADS; ++h){
    const float* qrow = q + ((((size_t)n * TDIM + qt) * LDIM + 1 + mrow) * HEADS + h) * DDIM;
    v16h a0 = ldA_f32(qrow, 0,  hi, 0.125f);   // scale = 1/sqrt(64)
    v16h a1 = ldA_f32(qrow, 32, hi, 0.125f);

    v8f acc[13];
#pragma unroll
    for (int j = 0; j < 13; ++j)
#pragma unroll
      for (int r = 0; r < 8; ++r) acc[j][r] = 0.0f;

#pragma unroll
    for (int j = 0; j < 13; ++j){
      int krow = j * 16 + m15;
      if (krow > QV - 1) krow = QV - 1;        // garbage cols masked in softmax
      const float* krp = k + ((((size_t)n * TDIM + kt) * LDIM + 1 + krow) * HEADS + h) * DDIM;
      v16h b0 = ldB_f32(krp, 0,  hi);
      v16h b1 = ldB_f32(krp, 32, hi);
      acc[j] = __builtin_amdgcn_wmma_f32_16x16x32_f16(false, a0, false, b0, (short)0, acc[j], false, false);
      acc[j] = __builtin_amdgcn_wmma_f32_16x16x32_f16(false, a1, false, b1, (short)0, acc[j], false, false);
    }

    // dump S strip: C layout -> row = r + hi*8, col = j*16 + (lane&15)
#pragma unroll
    for (int j = 0; j < 13; ++j)
#pragma unroll
      for (int r = 0; r < 8; ++r)
        sS[r + hi * 8][j * 16 + m15] = acc[j][r];
    __syncthreads();

    // row softmax over keys; lane pair (l, l+16) splits each row's 208 cols
    const int row = m15;
    const int cb  = hi * 104;
    float mx = -3.0e38f;
    for (int cc = 0; cc < 104; ++cc){
      int col = cb + cc;
      if (col < QV){ float v = sS[row][col]; mx = v > mx ? v : mx; }
    }
    { float o = __shfl_xor(mx, 16, 32); mx = o > mx ? o : mx; }
    float sum = 0.0f;
    for (int cc = 0; cc < 104; ++cc){
      int col = cb + cc;
      if (col < QV){ float e = __expf(sS[row][col] - mx); sS[row][col] = e; sum += e; }
    }
    sum += __shfl_xor(sum, 16, 32);
    float sc = 1.0f / (sum * (float)HEADS);    // head-mean folded in
    for (int cc = 0; cc < 104; ++cc){
      int col = cb + cc;
      if (col < QV) pacc[row][col] += sS[row][col] * sc;
    }
    __syncthreads();
  }

  // emit f16 P[half][q][nt][kpad]; pad cols (>=196) stay exactly 0
  const int nt = n * 7 + tp;
  for (int i = lane; i < 16 * KPAD; i += 32){
    int r = i / KPAD, col = i % KPAD;
    int qg = qt_i * 16 + r;
    if (qg < QV)
      pws[(((size_t)hf * QV + qg) * NT + nt) * KPAD + col] = (h16)pacc[r][col];
  }
}

// ---------------------------------------------------------------------------
// Kernel 2: out[nt, c] = P[hf,q] @ w[idx[q,:]] ; combine shifted halves
// grid (196 q, 12 col-tiles), 64 threads (2 waves, 32 cols each)
// ---------------------------------------------------------------------------
__global__ __launch_bounds__(64)
void EVL_attn_out(const h16* __restrict__ pws, const float* __restrict__ w1,
                  const float* __restrict__ w2, const int* __restrict__ idx,
                  float* __restrict__ out){
  const int qi  = blockIdx.x;
  const int c0  = blockIdx.y * CBLK;
  const int tid = threadIdx.x;
  const int lane = tid & 31;
  const int wv   = tid >> 5;
  const int m15  = lane & 15;
  const int hi   = lane >> 4;

  __shared__ h16  wq[CBLK][WKPAD];   // transposed gathered weights: [c][k]
  __shared__ float r0[NT][CBLK];     // half-0 results staged for combining
  __shared__ int  irow[KPAD];

  for (int kk = tid; kk < KPAD; kk += 64) irow[kk] = (kk < QV) ? idx[qi * QV + kk] : -1;
  __syncthreads();

  for (int hf = 0; hf < 2; ++hf){
    const float* w = hf ? w2 : w1;
    // gather w[idx[q,k]][c0+tid] -> wq[tid][k] (f16), zero the K pad
    for (int kk = 0; kk < KPAD; ++kk){
      int rw = irow[kk];
      float v = (rw >= 0) ? w[(size_t)rw * CDIM + c0 + tid] : 0.0f;
      wq[tid][kk] = (h16)v;
    }
    __syncthreads();

    v8f acc[7][2];
#pragma unroll
    for (int mt = 0; mt < 7; ++mt)
#pragma unroll
      for (int j = 0; j < 2; ++j)
#pragma unroll
        for (int r = 0; r < 8; ++r) acc[mt][j][r] = 0.0f;

    const h16* pbase = pws + ((size_t)hf * QV + qi) * NT * KPAD;
    for (int ks = 0; ks < 7; ++ks){
      v16h b[2];
#pragma unroll
      for (int j = 0; j < 2; ++j){
        const h16* bp = &wq[wv * 32 + j * 16 + m15][ks * 32 + hi * 16];
        v8h lo = *(const v8h*)bp;
        v8h hv = *(const v8h*)(bp + 8);
        v16h bb;
#pragma unroll
        for (int i = 0; i < 8; ++i){ bb[i] = lo[i]; bb[8 + i] = hv[i]; }
        b[j] = bb;
      }
#pragma unroll
      for (int mt = 0; mt < 7; ++mt){
        const h16* ap = pbase + (size_t)(mt * 16 + m15) * KPAD + ks * 32 + hi * 16;
        v8h lo = *(const v8h*)ap;
        v8h hv = *(const v8h*)(ap + 8);
        v16h aa;
#pragma unroll
        for (int i = 0; i < 8; ++i){ aa[i] = lo[i]; aa[8 + i] = hv[i]; }
        acc[mt][0] = __builtin_amdgcn_wmma_f32_16x16x32_f16(false, aa, false, b[0], (short)0, acc[mt][0], false, false);
        acc[mt][1] = __builtin_amdgcn_wmma_f32_16x16x32_f16(false, aa, false, b[1], (short)0, acc[mt][1], false, false);
      }
    }
    __syncthreads();

    if (hf == 0){
#pragma unroll
      for (int mt = 0; mt < 7; ++mt)
#pragma unroll
        for (int j = 0; j < 2; ++j)
#pragma unroll
          for (int r = 0; r < 8; ++r)
            r0[mt * 16 + r + hi * 8][wv * 32 + j * 16 + m15] = acc[mt][j][r];
      __syncthreads();
    } else {
      // out[n,t] = R0[n,t-1] + R1[n,t] for t=0..6 (t=0 has no R0 term)
#pragma unroll
      for (int mt = 0; mt < 7; ++mt)
#pragma unroll
        for (int j = 0; j < 2; ++j){
          int colL = wv * 32 + j * 16 + m15;
#pragma unroll
          for (int r = 0; r < 8; ++r){
            int ntr = mt * 16 + r + hi * 8;
            int nn = ntr / 7, t = ntr % 7;
            float v = acc[mt][j][r] + (t >= 1 ? r0[ntr - 1][colL] : 0.0f);
            out[(((size_t)nn * TDIM + t) * LDIM + 1 + qi) * CDIM + c0 + colL] = v;
          }
        }
      // t = 7: only R0[n, t'=6]
      for (int i = tid; i < NBATCH * CBLK; i += 64){
        int nn = i / CBLK, colL = i % CBLK;
        out[(((size_t)nn * TDIM + 7) * LDIM + 1 + qi) * CDIM + c0 + colL] = r0[nn * 7 + 6][colL];
      }
    }
  }
}

// zero the CLS (l = 0) plane of the output
__global__ void EVL_zero_cls(float* __restrict__ out){
  int i = blockIdx.x * blockDim.x + threadIdx.x;
  if (i < NBATCH * TDIM * CDIM){
    int ntt = i / CDIM, c = i % CDIM;
    out[(size_t)ntt * LDIM * CDIM + c] = 0.0f;
  }
}

extern "C" void kernel_launch(void* const* d_in, const int* in_sizes, int n_in,
                              void* d_out, int out_size, void* d_ws, size_t ws_size,
                              hipStream_t stream){
  const float* q  = (const float*)d_in[0];
  const float* k  = (const float*)d_in[1];
  const float* w1 = (const float*)d_in[2];
  const float* w2 = (const float*)d_in[3];
  const int*   idx = (const int*)d_in[4];
  float* out = (float*)d_out;
  h16* pws = (h16*)d_ws;   // needs 2*196*112*224 f16 ~ 19.7 MB

  EVL_zero_cls<<<dim3((NBATCH * TDIM * CDIM + 255) / 256), dim3(256), 0, stream>>>(out);
  EVL_attn_probs<<<dim3(13, 14, NBATCH), dim3(32), 0, stream>>>(q, k, pws);
  EVL_attn_out<<<dim3(QV, CDIM / CBLK), dim3(64), 0, stream>>>(pws, w1, w2, idx, out);
}
